// EASLayer_MLP_62551903699397
// MI455X (gfx1250) — matrix operand order repaired and checked by reading, not compiled, not run on hardware
//
#include <hip/hip_runtime.h>
#include <hip/hip_bf16.h>
#include <math.h>

typedef __attribute__((ext_vector_type(16))) _Float16 v16h;
typedef __attribute__((ext_vector_type(8)))  _Float16 v8h;
typedef __attribute__((ext_vector_type(8)))  float    v8f;

#define M_TOTAL 4928   // 64*77
#define D_DIM   768
#define NR      4096   // N*R = 1024*4

__device__ __forceinline__ float gelu_exact(float x) {
    return 0.5f * x * (1.0f + erff(x * 0.70710678118654752440f));
}

__device__ __forceinline__ v8f wmma16(v16h a, v16h b, v8f c) {
    return __builtin_amdgcn_wmma_f32_16x16x32_f16(false, a, false, b, (short)0, c, false, false);
}

// A fragment: row-major A [*, ld], 16x32 tile at (row0, k0).
// ISA layout: lanes 0-15 = M rows (K 0..7 in v0-3, 16..23 in v4-7),
//             lanes 16-31 = same rows (K 8..15, 24..31).
__device__ __forceinline__ v16h load_a_frag(const _Float16* __restrict__ A, int ld,
                                            int row0, int k0, int lane) {
    const int m = lane & 15, half = lane >> 4;
    const _Float16* p = A + (size_t)(row0 + m) * ld + k0 + half * 8;
    union { v16h v; v8h h[2]; } u;
    u.h[0] = *(const v8h*)(p);        // 16B: K chunk 0 (or 8)
    u.h[1] = *(const v8h*)(p + 16);   // 16B: K chunk 16 (or 24)
    return u.v;
}

// B fragment: column-major B (column stride ld), 32x16 tile at (k0, col0).
// ISA layout: lanes 0-15 = N cols holding K 0..7,16..23, lanes 16-31 = K 8..15,24..31.
__device__ __forceinline__ v16h load_b_frag(const _Float16* __restrict__ Bc, int ld,
                                            int col0, int k0, int lane) {
    const int n = lane & 15, half = lane >> 4;
    const _Float16* p = Bc + (size_t)(col0 + n) * ld + k0 + half * 8;
    union { v16h v; v8h h[2]; } u;
    u.h[0] = *(const v8h*)(p);
    u.h[1] = *(const v8h*)(p + 16);
    return u.v;
}

// ---------------- prep kernels ----------------

// straight f32 -> f16 convert (used for x and for W_org: row d of W_org is
// exactly col-major column d of B for X @ W_org^T)
__global__ void cvt_f32_f16(const float* __restrict__ src, _Float16* __restrict__ dst, int n) {
    int i = blockIdx.x * blockDim.x + threadIdx.x;
    if (i < n) dst[i] = (_Float16)src[i];
}

// w1[n,d,r] (flat n*3072 + d*4 + r) -> col-major W1b[c*768 + d], c = n*4+r
__global__ void cvt_w1(const float* __restrict__ w1, _Float16* __restrict__ W1b) {
    int i = blockIdx.x * blockDim.x + threadIdx.x;       // over 4096*768
    if (i >= NR * D_DIM) return;
    int c = i / D_DIM, d = i - c * D_DIM;
    W1b[i] = (_Float16)w1[(size_t)(c >> 2) * 3072 + d * 4 + (c & 3)];
}

// w_out flat is row-major [4096, 768]; build col-major WoutB[d*4096 + c]
__global__ void cvt_wout(const float* __restrict__ w_out, _Float16* __restrict__ WoutB) {
    int i = blockIdx.x * blockDim.x + threadIdx.x;       // over 768*4096
    if (i >= NR * D_DIM) return;
    int d = i / NR, c = i - d * NR;
    WoutB[i] = (_Float16)w_out[(size_t)c * D_DIM + d];
}

// bias_sum[d] = sum_n b_out[n, d]
__global__ void bias_reduce(const float* __restrict__ b_out, float* __restrict__ bias_sum) {
    int d = blockIdx.x * blockDim.x + threadIdx.x;
    if (d >= D_DIM) return;
    float s = 0.0f;
    for (int n = 0; n < 1024; ++n) s += b_out[(size_t)n * D_DIM + d];
    bias_sum[d] = s;
}

// ---------------- GEMM 1: H2 = act-chain(X @ W1cat) ----------------
// grid (77, 8), block 256 = 8 waves; wave tile 64x64 (4 M-tiles x 4 N-tiles)
__global__ __launch_bounds__(256)
void gemm1_act(const _Float16* __restrict__ Xh, const _Float16* __restrict__ W1b,
               const float* __restrict__ b1, const float* __restrict__ midw,
               const float* __restrict__ midb, _Float16* __restrict__ H2) {
    const int lane = threadIdx.x & 31;
    const int wave = threadIdx.x >> 5;
    const int row0 = blockIdx.x * 64;
    const int col0 = blockIdx.y * 512 + wave * 64;

    v8f acc[4][4] = {};
    for (int k0 = 0; k0 < D_DIM; k0 += 32) {
        v16h a[4], b[4];
#pragma unroll
        for (int i = 0; i < 4; ++i) a[i] = load_a_frag(Xh, D_DIM, row0 + 16 * i, k0, lane);
#pragma unroll
        for (int j = 0; j < 4; ++j) b[j] = load_b_frag(W1b, D_DIM, col0 + 16 * j, k0, lane);
#pragma unroll
        for (int i = 0; i < 4; ++i)
#pragma unroll
            for (int j = 0; j < 4; ++j) acc[i][j] = wmma16(a[i], b[j], acc[i][j]);
    }

    // epilogue: bias -> gelu -> diag(mid_w)*h + mid_b -> gelu -> f16 store
    const int n = lane & 15, half = lane >> 4;
#pragma unroll
    for (int j = 0; j < 4; ++j) {
        const int c = col0 + 16 * j + n;
        const float b1c = b1[c];
        const float md  = midw[(c >> 2) * 16 + (c & 3) * 5];  // mid_w[n,r,r]
        const float mb  = midb[c];
#pragma unroll
        for (int i = 0; i < 4; ++i) {
            const int mrow = row0 + 16 * i + half * 8;
#pragma unroll
            for (int r = 0; r < 8; ++r) {
                float h = acc[i][j][r] + b1c;
                h = gelu_exact(h);
                h = h * md + mb;
                h = gelu_exact(h);
                H2[(size_t)(mrow + r) * NR + c] = (_Float16)h;
            }
        }
    }
}

// ---------------- GEMM 2: out = X@W_org^T + 0.25*(H2@Wout + sum b_out) + b_org ----
// grid (77, 3), block 128 = 4 waves; wave tile 64x64
__global__ __launch_bounds__(128)
void gemm2_fused(const _Float16* __restrict__ H2, const _Float16* __restrict__ WoutB,
                 const _Float16* __restrict__ Xh, const _Float16* __restrict__ WorgB,
                 const float* __restrict__ b_org, const float* __restrict__ bias_sum,
                 float* __restrict__ out) {
    const int lane = threadIdx.x & 31;
    const int wave = threadIdx.x >> 5;
    const int row0 = blockIdx.x * 64;
    const int col0 = blockIdx.y * 256 + wave * 64;

    v8f acc[4][4] = {};
    // pass 1: UP = H2 @ Wout  (K = 4096)
    for (int k0 = 0; k0 < NR; k0 += 32) {
        v16h a[4], b[4];
#pragma unroll
        for (int i = 0; i < 4; ++i) a[i] = load_a_frag(H2, NR, row0 + 16 * i, k0, lane);
#pragma unroll
        for (int j = 0; j < 4; ++j) b[j] = load_b_frag(WoutB, NR, col0 + 16 * j, k0, lane);
#pragma unroll
        for (int i = 0; i < 4; ++i)
#pragma unroll
            for (int j = 0; j < 4; ++j) acc[i][j] = wmma16(a[i], b[j], acc[i][j]);
    }
    // scale UP by MULTIPLIER*SCALE = 0.25 before accumulating ORG on top
#pragma unroll
    for (int i = 0; i < 4; ++i)
#pragma unroll
        for (int j = 0; j < 4; ++j)
#pragma unroll
            for (int r = 0; r < 8; ++r) acc[i][j][r] *= 0.25f;

    // pass 2: ORG = X @ W_org^T  (K = 768; WorgB col d = W_org row d)
    for (int k0 = 0; k0 < D_DIM; k0 += 32) {
        v16h a[4], b[4];
#pragma unroll
        for (int i = 0; i < 4; ++i) a[i] = load_a_frag(Xh, D_DIM, row0 + 16 * i, k0, lane);
#pragma unroll
        for (int j = 0; j < 4; ++j) b[j] = load_b_frag(WorgB, D_DIM, col0 + 16 * j, k0, lane);
#pragma unroll
        for (int i = 0; i < 4; ++i)
#pragma unroll
            for (int j = 0; j < 4; ++j) acc[i][j] = wmma16(a[i], b[j], acc[i][j]);
    }

    const int n = lane & 15, half = lane >> 4;
#pragma unroll
    for (int j = 0; j < 4; ++j) {
        const int c = col0 + 16 * j + n;
        const float bias = b_org[c] + 0.25f * bias_sum[c];
#pragma unroll
        for (int i = 0; i < 4; ++i) {
            const int mrow = row0 + 16 * i + half * 8;
#pragma unroll
            for (int r = 0; r < 8; ++r)
                out[(size_t)(mrow + r) * D_DIM + c] = acc[i][j][r] + bias;
        }
    }
}

extern "C" void kernel_launch(void* const* d_in, const int* in_sizes, int n_in,
                              void* d_out, int out_size, void* d_ws, size_t ws_size,
                              hipStream_t stream) {
    const float* x     = (const float*)d_in[0];
    const float* W_org = (const float*)d_in[1];
    const float* b_org = (const float*)d_in[2];
    const float* w1    = (const float*)d_in[3];
    const float* b1    = (const float*)d_in[4];
    const float* mid_w = (const float*)d_in[5];
    const float* mid_b = (const float*)d_in[6];
    const float* w_out = (const float*)d_in[7];
    const float* b_out = (const float*)d_in[8];
    float* out = (float*)d_out;

    // workspace carve-up (256B aligned)
    char* ws = (char*)d_ws;
    size_t off = 0;
    auto take = [&](size_t bytes) { char* p = ws + off; off = (off + bytes + 255) & ~(size_t)255; return p; };
    _Float16* Xh     = (_Float16*)take((size_t)M_TOTAL * D_DIM * 2); // 7.6 MB
    _Float16* W1b    = (_Float16*)take((size_t)NR * D_DIM * 2);      // 6.3 MB (col-major)
    _Float16* WoutB  = (_Float16*)take((size_t)NR * D_DIM * 2);      // 6.3 MB (col-major)
    _Float16* WorgB  = (_Float16*)take((size_t)D_DIM * D_DIM * 2);   // 1.2 MB
    _Float16* H2     = (_Float16*)take((size_t)M_TOTAL * NR * 2);    // 40.4 MB
    float*    bsum   = (float*)take((size_t)D_DIM * 4);

    const int TB = 256;
    // prep
    {
        int n = M_TOTAL * D_DIM;
        cvt_f32_f16<<<(n + TB - 1) / TB, TB, 0, stream>>>(x, Xh, n);
    }
    {
        int n = D_DIM * D_DIM;
        cvt_f32_f16<<<(n + TB - 1) / TB, TB, 0, stream>>>(W_org, WorgB, n);
    }
    {
        int n = NR * D_DIM;
        cvt_w1<<<(n + TB - 1) / TB, TB, 0, stream>>>(w1, W1b);
        cvt_wout<<<(n + TB - 1) / TB, TB, 0, stream>>>(w_out, WoutB);
    }
    bias_reduce<<<(D_DIM + TB - 1) / TB, TB, 0, stream>>>(b_out, bsum);

    // GEMM1 + activation chain -> H2 (f16)
    gemm1_act<<<dim3(M_TOTAL / 64, NR / 512), 256, 0, stream>>>(Xh, W1b, b1, mid_w, mid_b, H2);

    // GEMM2 (UP) fused with GEMM3 (ORG) and all biases -> out (f32)
    gemm2_fused<<<dim3(M_TOTAL / 64, D_DIM / 256), 128, 0, stream>>>(
        H2, WoutB, Xh, WorgB, b_org, bsum, out);
}
